// GNNEncoder_77429670412868
// MI455X (gfx1250) — compile-verified, matrix operand order checked by reading, not run on hardware
//
#include <hip/hip_runtime.h>
#include <hip/hip_bf16.h>
#include <math.h>

// ---------------- problem constants ----------------
static const int kN   = 50000;   // nodes  (16*3125)
static const int kE   = 500000;  // edges  (16*31250)
static const int kEMB = 50;      // embedding dim (pad -> 64)
static const int kED  = 28;      // edge feature dim (pad -> 32)
static const int kC1  = 256;     // H*HID = 4*64
static const int kH   = 4;
static const int kOUT = 50;      // conv2 out (pad -> 64)

typedef __attribute__((ext_vector_type(16))) __bf16 v16bf;
typedef __attribute__((ext_vector_type(8)))  float  v8f;

// ---------------- helpers ----------------
__device__ __forceinline__ unsigned short f2bf(float f) {
  unsigned u = __float_as_uint(f);
  u += 0x7FFFu + ((u >> 16) & 1u);           // round-to-nearest-even
  return (unsigned short)(u >> 16);
}

// WMMA 16x16x32 bf16 fragment loader.
// p points at (row-or-col start + k-chunk base) in a bf16 (ushort) buffer whose
// K dimension is contiguous.  Per ISA 7.12.2: lanes 0-15 hold K=0..7,16..23,
// lanes 16-31 hold K=8..15,24..31; 2 bf16 per VGPR dword.
__device__ __forceinline__ v16bf load_frag(const unsigned short* p, int hi) {
  union { v16bf v; unsigned u[8]; } f;
#pragma unroll
  for (int i = 0; i < 8; ++i) {
    int k = ((i >> 2) << 4) + (hi << 3) + ((i & 3) << 1);
    f.u[i] = *reinterpret_cast<const unsigned*>(p + k);
  }
  return f.v;
}

__device__ __forceinline__ unsigned fkey(float f) {
  unsigned b = __float_as_uint(f);
  return (b & 0x80000000u) ? ~b : (b | 0x80000000u);
}
__device__ __forceinline__ float funkey(unsigned k) {
  return __uint_as_float((k & 0x80000000u) ? (k ^ 0x80000000u) : ~k);
}

// ---------------- cast / pad kernels ----------------
__global__ void k_cast_h(const int* __restrict__ xidx, const float* __restrict__ emb,
                         unsigned short* __restrict__ hb) {
  int i = blockIdx.x * blockDim.x + threadIdx.x;
  if (i >= kN * 64) return;
  int n = i >> 6, c = i & 63;
  float v = (c < kEMB) ? emb[n < kN ? (long)xidx[n] * kEMB + c : 0] : 0.f;
  hb[i] = f2bf(v);
}

__global__ void k_cast_ea(const float* __restrict__ ea, unsigned short* __restrict__ eb) {
  int i = blockIdx.x * blockDim.x + threadIdx.x;
  if (i >= kE * 32) return;
  int e = i >> 5, c = i & 31;
  float v = (c < kED) ? ea[(long)e * kED + c] : 0.f;
  eb[i] = f2bf(v);
}

// W[K,NN] f32 -> WT[NNpad,Kpad] bf16 (transposed, zero padded)
__global__ void k_cast_wT(const float* __restrict__ W, int K, int NN,
                          unsigned short* __restrict__ WT, int Kpad, int NNpad) {
  int i = blockIdx.x * blockDim.x + threadIdx.x;
  if (i >= NNpad * Kpad) return;
  int n = i / Kpad, k = i - n * Kpad;
  float v = (k < K && n < NN) ? W[(long)k * NN + n] : 0.f;
  WT[i] = f2bf(v);
}

__global__ void k_fill_u32(unsigned* p, unsigned v, int n) {
  int i = blockIdx.x * blockDim.x + threadIdx.x;
  if (i < n) p[i] = v;
}
__global__ void k_fill_f32(float* p, float v, int n) {
  int i = blockIdx.x * blockDim.x + threadIdx.x;
  if (i < n) p[i] = v;
}
__global__ void k_out_init(float* out, const float* __restrict__ bias, int nvalid, int ld, int n) {
  int i = blockIdx.x * blockDim.x + threadIdx.x;
  if (i >= n) return;
  int c = i % ld;
  out[i] = (c < nvalid) ? bias[c] : 0.f;
}

// ---------------- WMMA node GEMM: C[M,N] = A_bf16[M,K] @ WT_bf16[N,K]^T + bias ----------------
__global__ void k_gemm(const unsigned short* __restrict__ A, int lda,
                       const unsigned short* __restrict__ Wt, int ldk, int K,
                       const float* __restrict__ bias, int nbias,
                       float* __restrict__ C, int ldc, int mtiles, int ntiles) {
  int wave = (blockIdx.x * blockDim.x + threadIdx.x) >> 5;
  if (wave >= mtiles * ntiles) return;          // wave-uniform: EXEC stays all-1 for WMMA
  int lane = threadIdx.x & 31;
  int mt = wave / ntiles, nt = wave - mt * ntiles;
  int l16 = lane & 15, hi = lane >> 4;
  const unsigned short* arow = A  + (long)(mt * 16 + l16) * lda;
  const unsigned short* bcol = Wt + (long)(nt * 16 + l16) * ldk;
  v8f c = {};
  for (int k = 0; k < K; k += 32) {
    v16bf a = load_frag(arow + k, hi);
    v16bf b = load_frag(bcol + k, hi);
    c = __builtin_amdgcn_wmma_f32_16x16x32_bf16(false, a, false, b, (short)0, c, false, false);
  }
  int col = nt * 16 + l16;
  float bv = (col < nbias) ? bias[col] : 0.f;
#pragma unroll
  for (int r = 0; r < 8; ++r) {
    int row = mt * 16 + r + hi * 8;
    C[(long)row * ldc + col] = c[r] + bv;
  }
}

// ---------------- fused WMMA edge-logit kernel ----------------
// One wave = 16 edges.  xe tile computed by WMMA (never materialized to HBM),
// gathered xl[src]/xr[dst] added, leaky-relu, dot with att -> logits[e,h].
__global__ void k_edge_logits(const unsigned short* __restrict__ EA,   // [E,32] bf16
                              const int* __restrict__ src, const int* __restrict__ dst,
                              const float* __restrict__ XL, const float* __restrict__ XR,
                              const unsigned short* __restrict__ WeT,  // [heads*64,32] bf16
                              const float* __restrict__ att, int attvalid,
                              float* __restrict__ logits, int heads, int etiles) {
  int wave = (blockIdx.x * blockDim.x + threadIdx.x) >> 5;
  if (wave >= etiles) return;
  int lane = threadIdx.x & 31;
  int l16 = lane & 15, hi = lane >> 4;
  int e0 = wave * 16;
  int ldx = heads * 64;
  int s[8], d[8];
#pragma unroll
  for (int r = 0; r < 8; ++r) {
    int m = r + hi * 8;
    s[r] = src[e0 + m];
    d[r] = dst[e0 + m];
  }
  v16bf a = load_frag(EA + (long)(e0 + l16) * 32, hi);   // 16 edges x K=32
  for (int h = 0; h < heads; ++h) {
    float acc[8];
#pragma unroll
    for (int r = 0; r < 8; ++r) acc[r] = 0.f;
    for (int st = 0; st < 4; ++st) {                      // 4 x 16 cols = one head
      int col = h * 64 + st * 16 + l16;
      v16bf b = load_frag(WeT + (long)col * 32, hi);
      v8f c = {};
      c = __builtin_amdgcn_wmma_f32_16x16x32_bf16(false, a, false, b, (short)0, c, false, false);
      int cc = st * 16 + l16;
      float av = (cc < attvalid) ? att[h * attvalid + cc] : 0.f;
#pragma unroll
      for (int r = 0; r < 8; ++r) {
        float v = c[r] + XL[(long)s[r] * ldx + col] + XR[(long)d[r] * ldx + col];
        v = (v > 0.f) ? v : 0.2f * v;                     // leaky_relu 0.2
        acc[r] += v * av;
      }
    }
#pragma unroll
    for (int r = 0; r < 8; ++r) {                         // reduce over 16 cols
      float t = acc[r];
      t += __shfl_xor(t, 1); t += __shfl_xor(t, 2);
      t += __shfl_xor(t, 4); t += __shfl_xor(t, 8);
      acc[r] = t;
    }
    if (l16 == 0) {
#pragma unroll
      for (int r = 0; r < 8; ++r) {
        int m = r + hi * 8;
        logits[(long)(e0 + m) * heads + h] = acc[r];
      }
    }
  }
}

// ---------------- edge softmax (segment max / exp+sum) ----------------
__global__ void k_seg_max(const float* __restrict__ logits, const int* __restrict__ dst,
                          unsigned* amax, int heads, int total) {
  int i = blockIdx.x * blockDim.x + threadIdx.x;
  if (i >= total) return;
  int e = i / heads, h = i - e * heads;
  atomicMax(&amax[(long)dst[e] * heads + h], fkey(logits[i]));
}

__global__ void k_seg_exp(float* logits, const int* __restrict__ dst,
                          const unsigned* __restrict__ amax, float* denom, int heads, int total) {
  int i = blockIdx.x * blockDim.x + threadIdx.x;
  if (i >= total) return;
  int e = i / heads, h = i - e * heads;
  float mx = funkey(amax[(long)dst[e] * heads + h]);
  float ex = __expf(logits[i] - mx);
  logits[i] = ex;
  atomicAdd(&denom[(long)dst[e] * heads + h], ex);
}

// out[dst] += xl[src] * alpha   (alpha = ex / (denom + eps))
__global__ void k_agg(const float* __restrict__ ex, const float* __restrict__ denom,
                      const int* __restrict__ src, const int* __restrict__ dst,
                      const float* __restrict__ XL, float* out, int heads, int ld, long total) {
  long i = (long)blockIdx.x * blockDim.x + threadIdx.x;
  if (i >= total) return;
  int e = (int)(i / ld);
  int col = (int)(i - (long)e * ld);
  int h = col >> 6;
  int de = dst[e];
  float alpha = ex[(long)e * heads + h] / (denom[(long)de * heads + h] + 1e-16f);
  atomicAdd(&out[(long)de * ld + col], XL[(long)src[e] * ld + col] * alpha);
}

// ---------------- graph LayerNorm (scalar mean/std over whole tensor) ----------------
__global__ void k_stats(const float* __restrict__ x, int nrows, int ncols, int ldc, float* stats) {
  long total = (long)nrows * ncols;
  float s = 0.f, s2 = 0.f;
  for (long i = (long)blockIdx.x * blockDim.x + threadIdx.x; i < total;
       i += (long)gridDim.x * blockDim.x) {
    int n = (int)(i / ncols), c = (int)(i - (long)n * ncols);
    float v = x[(long)n * ldc + c];
    s += v; s2 += v * v;
  }
  for (int off = 16; off; off >>= 1) { s += __shfl_xor(s, off); s2 += __shfl_xor(s2, off); }
  if ((threadIdx.x & 31) == 0) { atomicAdd(&stats[0], s); atomicAdd(&stats[1], s2); }
}

__global__ void k_ln_relu_bf(const float* __restrict__ x, const float* __restrict__ w,
                             const float* __restrict__ b, const float* __restrict__ stats,
                             float cnt, unsigned short* out, int n, int ncols) {
  int i = blockIdx.x * blockDim.x + threadIdx.x;
  if (i >= n) return;
  int c = i % ncols;
  float mean = stats[0] / cnt;
  float var  = stats[1] / cnt - mean * mean;
  float sd   = sqrtf(fmaxf(var, 0.f));
  float y = (x[i] - mean) / (sd + 1e-5f) * w[c] + b[c];
  out[i] = f2bf(fmaxf(y, 0.f));
}

__global__ void k_ln_out(const float* __restrict__ x, int ldc, const float* __restrict__ w,
                         const float* __restrict__ b, const float* __restrict__ stats,
                         float cnt, float* out, int nrows, int ncols) {
  int i = blockIdx.x * blockDim.x + threadIdx.x;
  if (i >= nrows * ncols) return;
  int n = i / ncols, c = i - n * ncols;
  float mean = stats[0] / cnt;
  float var  = stats[1] / cnt - mean * mean;
  float sd   = sqrtf(fmaxf(var, 0.f));
  out[i] = (x[(long)n * ldc + c] - mean) / (sd + 1e-5f) * w[c] + b[c];
}

// ---------------- launch ----------------
static inline int cdiv(long a, long b) { return (int)((a + b - 1) / b); }

extern "C" void kernel_launch(void* const* d_in, const int* in_sizes, int n_in,
                              void* d_out, int out_size, void* d_ws, size_t ws_size,
                              hipStream_t stream) {
  (void)in_sizes; (void)n_in; (void)out_size; (void)ws_size;
  const int*   xidx = (const int*)d_in[0];
  const int*   src  = (const int*)d_in[1];
  const int*   dst  = src + kE;
  const float* ea   = (const float*)d_in[2];
  const float* emb  = (const float*)d_in[3];
  const float *Wl1 = (const float*)d_in[4],  *bl1 = (const float*)d_in[5];
  const float *Wr1 = (const float*)d_in[6],  *br1 = (const float*)d_in[7];
  const float *We1 = (const float*)d_in[8],  *att1 = (const float*)d_in[9];
  const float *bias1 = (const float*)d_in[10];
  const float *ln1w = (const float*)d_in[11], *ln1b = (const float*)d_in[12];
  const float *Wl2 = (const float*)d_in[13], *bl2 = (const float*)d_in[14];
  const float *Wr2 = (const float*)d_in[15], *br2 = (const float*)d_in[16];
  const float *We2 = (const float*)d_in[17], *att2 = (const float*)d_in[18];
  const float *bias2 = (const float*)d_in[19];
  const float *ln2w = (const float*)d_in[20], *ln2b = (const float*)d_in[21];
  float* out = (float*)d_out;

  // workspace carve-out
  char* w = (char*)d_ws;
  size_t cur = 0;
  auto alloc = [&](size_t bytes) { void* p = w + cur; cur = (cur + bytes + 255) & ~(size_t)255; return p; };
  unsigned short* h_bf  = (unsigned short*)alloc((size_t)kN * 64 * 2);
  unsigned short* ea_bf = (unsigned short*)alloc((size_t)kE * 32 * 2);
  unsigned short* WlT1  = (unsigned short*)alloc((size_t)256 * 64 * 2);
  unsigned short* WrT1  = (unsigned short*)alloc((size_t)256 * 64 * 2);
  unsigned short* WeT1  = (unsigned short*)alloc((size_t)256 * 32 * 2);
  unsigned short* WlT2  = (unsigned short*)alloc((size_t)64 * 256 * 2);
  unsigned short* WrT2  = (unsigned short*)alloc((size_t)64 * 256 * 2);
  unsigned short* WeT2  = (unsigned short*)alloc((size_t)64 * 32 * 2);
  float*    xl1    = (float*)alloc((size_t)kN * 256 * 4);
  float*    xr1    = (float*)alloc((size_t)kN * 256 * 4);
  float*    lg1    = (float*)alloc((size_t)kE * 4 * 4);
  unsigned* amax1  = (unsigned*)alloc((size_t)kN * 4 * 4);
  float*    den1   = (float*)alloc((size_t)kN * 4 * 4);
  float*    out1   = (float*)alloc((size_t)kN * 256 * 4);
  unsigned short* h2_bf = (unsigned short*)alloc((size_t)kN * 256 * 2);
  float*    xl2    = (float*)alloc((size_t)kN * 64 * 4);
  float*    xr2    = (float*)alloc((size_t)kN * 64 * 4);
  float*    lg2    = (float*)alloc((size_t)kE * 4);
  unsigned* amax2  = (unsigned*)alloc((size_t)kN * 4);
  float*    den2   = (float*)alloc((size_t)kN * 4);
  float*    out2   = (float*)alloc((size_t)kN * 64 * 4);
  float*    stats  = (float*)alloc(64);

  const int B = 256;
  // ---- casts / padding ----
  k_cast_h <<<cdiv((long)kN * 64, B), B, 0, stream>>>(xidx, emb, h_bf);
  k_cast_ea<<<cdiv((long)kE * 32, B), B, 0, stream>>>(ea, ea_bf);
  k_cast_wT<<<cdiv(256 * 64, B), B, 0, stream>>>(Wl1, kEMB, 256, WlT1, 64, 256);
  k_cast_wT<<<cdiv(256 * 64, B), B, 0, stream>>>(Wr1, kEMB, 256, WrT1, 64, 256);
  k_cast_wT<<<cdiv(256 * 32, B), B, 0, stream>>>(We1, kED, 256, WeT1, 32, 256);
  k_cast_wT<<<cdiv(64 * 256, B), B, 0, stream>>>(Wl2, 256, kOUT, WlT2, 256, 64);
  k_cast_wT<<<cdiv(64 * 256, B), B, 0, stream>>>(Wr2, 256, kOUT, WrT2, 256, 64);
  k_cast_wT<<<cdiv(64 * 32, B), B, 0, stream>>>(We2, kED, kOUT, WeT2, 32, 64);

  // ---- conv1 node GEMMs (WMMA): xl1/xr1 [N,256] ----
  {
    int waves = 3125 * 16, blocks = cdiv((long)waves * 32, B);
    k_gemm<<<blocks, B, 0, stream>>>(h_bf, 64, WlT1, 64, 64, bl1, 256, xl1, 256, 3125, 16);
    k_gemm<<<blocks, B, 0, stream>>>(h_bf, 64, WrT1, 64, 64, br1, 256, xr1, 256, 3125, 16);
  }
  // ---- conv1 softmax + aggregate ----
  k_fill_u32<<<cdiv(kN * 4, B), B, 0, stream>>>(amax1, 0u, kN * 4);
  k_fill_f32<<<cdiv(kN * 4, B), B, 0, stream>>>(den1, 0.f, kN * 4);
  k_out_init<<<cdiv((long)kN * 256, B), B, 0, stream>>>(out1, bias1, 256, 256, kN * 256);
  {
    int etiles = kE / 16, blocks = cdiv((long)etiles * 32, B);
    k_edge_logits<<<blocks, B, 0, stream>>>(ea_bf, src, dst, xl1, xr1, WeT1, att1, 64, lg1, 4, etiles);
  }
  k_seg_max<<<cdiv((long)kE * 4, B), B, 0, stream>>>(lg1, dst, amax1, 4, kE * 4);
  k_seg_exp<<<cdiv((long)kE * 4, B), B, 0, stream>>>(lg1, dst, amax1, den1, 4, kE * 4);
  k_agg<<<cdiv((long)kE * 256, B), B, 0, stream>>>(lg1, den1, src, dst, xl1, out1, 4, 256, (long)kE * 256);
  // ---- LN1 + ReLU -> bf16 h2 ----
  k_fill_f32<<<1, B, 0, stream>>>(stats, 0.f, 2);
  k_stats<<<1024, B, 0, stream>>>(out1, kN, 256, 256, stats);
  k_ln_relu_bf<<<cdiv((long)kN * 256, B), B, 0, stream>>>(out1, ln1w, ln1b, stats,
                                                          (float)((long)kN * 256), h2_bf, kN * 256, 256);
  // ---- conv2 node GEMMs: xl2/xr2 [N,64] (cols>=50 forced to 0 by padded WT/bias) ----
  {
    int waves = 3125 * 4, blocks = cdiv((long)waves * 32, B);
    k_gemm<<<blocks, B, 0, stream>>>(h2_bf, 256, WlT2, 256, 256, bl2, kOUT, xl2, 64, 3125, 4);
    k_gemm<<<blocks, B, 0, stream>>>(h2_bf, 256, WrT2, 256, 256, br2, kOUT, xr2, 64, 3125, 4);
  }
  // ---- conv2 softmax + aggregate ----
  k_fill_u32<<<cdiv(kN, B), B, 0, stream>>>(amax2, 0u, kN);
  k_fill_f32<<<cdiv(kN, B), B, 0, stream>>>(den2, 0.f, kN);
  k_out_init<<<cdiv((long)kN * 64, B), B, 0, stream>>>(out2, bias2, kOUT, 64, kN * 64);
  {
    int etiles = kE / 16, blocks = cdiv((long)etiles * 32, B);
    k_edge_logits<<<blocks, B, 0, stream>>>(ea_bf, src, dst, xl2, xr2, WeT2, att2, kOUT, lg2, 1, etiles);
  }
  k_seg_max<<<cdiv((long)kE, B), B, 0, stream>>>(lg2, dst, amax2, 1, kE);
  k_seg_exp<<<cdiv((long)kE, B), B, 0, stream>>>(lg2, dst, amax2, den2, 1, kE);
  k_agg<<<cdiv((long)kE * 64, B), B, 0, stream>>>(lg2, den2, src, dst, xl2, out2, 1, 64, (long)kE * 64);
  // ---- LN2 -> d_out [N,50] ----
  k_fill_f32<<<1, B, 0, stream>>>(stats, 0.f, 2);
  k_stats<<<1024, B, 0, stream>>>(out2, kN, kOUT, 64, stats);
  k_ln_out<<<cdiv((long)kN * kOUT, B), B, 0, stream>>>(out2, 64, ln2w, ln2b, stats,
                                                       (float)((long)kN * kOUT), out, kN, kOUT);
}
